// GCN_4801773437488
// MI455X (gfx1250) — matrix operand order repaired and checked by reading, not compiled, not run on hardware
//
#include <hip/hip_runtime.h>
#include <hip/hip_bf16.h>

typedef __attribute__((ext_vector_type(16))) _Float16 v16h;
typedef __attribute__((ext_vector_type(8)))  float    v8f;

#define NF 128
#define HID 64
#define NC 16

// Native (non-CAS) fp32 atomic add -> GLOBAL_ATOMIC_ADD_F32 on gfx1250.
__device__ __forceinline__ void atom_add_f32(float* p, float v) {
#if defined(__AMDGCN__)
    unsafeAtomicAdd(p, v);
#else
    atomicAdd(p, v);
#endif
}

// ---------------------------------------------------------------------------
// Degree / normalization
// ---------------------------------------------------------------------------
__global__ void deg_init_kernel(float* __restrict__ deg, int n) {
    int i = blockIdx.x * blockDim.x + threadIdx.x;
    if (i < n) deg[i] = 1.0f;              // self-loop contributes 1 to each node
}

__global__ void deg_edge_kernel(float* __restrict__ deg,
                                const int* __restrict__ dst, long e) {
    long i = (long)blockIdx.x * blockDim.x + threadIdx.x;
    if (i < e) atom_add_f32(&deg[dst[i]], 1.0f);
}

__global__ void dinv_kernel(float* __restrict__ dinv,
                            const float* __restrict__ deg, int n) {
    int i = blockIdx.x * blockDim.x + threadIdx.x;
    if (i < n) dinv[i] = rsqrtf(deg[i]);   // deg >= 1 always (self-loops)
}

// ---------------------------------------------------------------------------
// GEMM1: h1[N,64] = x[N,128] @ W1[128,64]   (f16 WMMA, f32 accumulate)
// One wave computes a 16x64 row-tile: 4 k-steps x 4 n-tiles = 16 WMMAs.
// ---------------------------------------------------------------------------
__global__ __launch_bounds__(256) void gemm1_wmma(const float* __restrict__ x,
                                                  const float* __restrict__ W,
                                                  float* __restrict__ h, int nrows) {
    __shared__ __align__(32) _Float16 Wt[HID * NF];   // Wt[n][k] = W[k][n], 16 KB
    for (int i = threadIdx.x; i < HID * NF; i += 256) {
        int n = i >> 7, k = i & 127;
        Wt[i] = (_Float16)W[k * HID + n];
    }
    __syncthreads();

    const int wave = threadIdx.x >> 5;
    const int lane = threadIdx.x & 31;
    const int m    = lane & 15;            // A row / B col / D col within tile
    const int g    = lane >> 4;            // lane half-group
    const int r0   = blockIdx.x * 128 + wave * 16;
    if (r0 >= nrows) return;

    v8f acc[4] = {};
    const float* xr = x + (long)(r0 + m) * NF;

#pragma unroll
    for (int ks = 0; ks < 4; ++ks) {
        const float* xk = xr + 32 * ks;
        v16h a;
#pragma unroll
        for (int j = 0; j < 8; ++j) {      // contiguous k order: k = 16g + s (matches B)
            a[j]     = (_Float16)xk[16 * g + j];
            a[8 + j] = (_Float16)xk[16 * g + 8 + j];
        }
#pragma unroll
        for (int nt = 0; nt < 4; ++nt) {
            const v16h b = *(const v16h*)(Wt + ((nt * 16 + m) * NF + 32 * ks + 16 * g));
            acc[nt] = __builtin_amdgcn_wmma_f32_16x16x32_f16(
                false, a, false, b, (short)0, acc[nt], false, false);
        }
    }

#pragma unroll
    for (int nt = 0; nt < 4; ++nt)
#pragma unroll
        for (int r = 0; r < 8; ++r)        // D layout: c[r] -> D[r + 8g][m]
            h[(long)(r0 + 8 * g + r) * HID + nt * 16 + m] = acc[nt][r];
}

// ---------------------------------------------------------------------------
// GEMM2: h2[N,16] = relu(out1)[N,64] @ W2[64,16]   (ReLU fused into A load)
// ---------------------------------------------------------------------------
__global__ __launch_bounds__(256) void gemm2_wmma(const float* __restrict__ a1,
                                                  const float* __restrict__ W,
                                                  float* __restrict__ h, int nrows) {
    __shared__ __align__(32) _Float16 Wt[NC * HID];   // Wt[n][k] = W[k][n], 2 KB
    for (int i = threadIdx.x; i < NC * HID; i += 256) {
        int n = i >> 6, k = i & 63;
        Wt[i] = (_Float16)W[k * NC + n];
    }
    __syncthreads();

    const int wave = threadIdx.x >> 5;
    const int lane = threadIdx.x & 31;
    const int m    = lane & 15;
    const int g    = lane >> 4;
    const int r0   = blockIdx.x * 128 + wave * 16;
    if (r0 >= nrows) return;

    v8f acc = {};
    const float* ar = a1 + (long)(r0 + m) * HID;

#pragma unroll
    for (int ks = 0; ks < 2; ++ks) {
        const float* ak = ar + 32 * ks;
        v16h a;
#pragma unroll
        for (int j = 0; j < 8; ++j) {
            float v0 = ak[16 * g + j];      v0 = v0 > 0.f ? v0 : 0.f;
            float v1 = ak[16 * g + 8 + j];  v1 = v1 > 0.f ? v1 : 0.f;
            a[j]     = (_Float16)v0;
            a[8 + j] = (_Float16)v1;
        }
        const v16h b = *(const v16h*)(Wt + (m * HID + 32 * ks + 16 * g));
        acc = __builtin_amdgcn_wmma_f32_16x16x32_f16(
            false, a, false, b, (short)0, acc, false, false);
    }

#pragma unroll
    for (int r = 0; r < 8; ++r)
        h[(long)(r0 + 8 * g + r) * NC + m] = acc[r];
}

// ---------------------------------------------------------------------------
// Aggregation: self-loop + bias init, then edge scatter-add (float4 payload)
// ---------------------------------------------------------------------------
template <int F>
__global__ void agg_init_kernel(float* __restrict__ out, const float* __restrict__ h,
                                const float* __restrict__ dinv,
                                const float* __restrict__ bias, long total) {
    long i = (long)blockIdx.x * blockDim.x + threadIdx.x;
    if (i >= total) return;
    long node = i / F;
    int  f    = (int)(i - node * F);
    float dn  = dinv[node];
    out[i] = h[i] * dn * dn + bias[f];     // self-loop msg + bias
}

template <int F>
__global__ void agg_edge_kernel(float* __restrict__ out, const float* __restrict__ h,
                                const float* __restrict__ dinv,
                                const int* __restrict__ src, const int* __restrict__ dst,
                                long e) {
    const int GPE = F / 4;                 // float4 groups per edge
    long i = (long)blockIdx.x * blockDim.x + threadIdx.x;
    long ed = i / GPE;
    if (ed >= e) return;
    int f4 = (int)(i - ed * GPE) * 4;
    int s = src[ed], d = dst[ed];
    float nrm = dinv[s] * dinv[d];
    const float4 v = *(const float4*)(h + (long)s * F + f4);
    float* o = out + (long)d * F + f4;
    atom_add_f32(o + 0, v.x * nrm);
    atom_add_f32(o + 1, v.y * nrm);
    atom_add_f32(o + 2, v.z * nrm);
    atom_add_f32(o + 3, v.w * nrm);
}

// ---------------------------------------------------------------------------
// Launch
// ---------------------------------------------------------------------------
extern "C" void kernel_launch(void* const* d_in, const int* in_sizes, int n_in,
                              void* d_out, int out_size, void* d_ws, size_t ws_size,
                              hipStream_t stream) {
    const float* x  = (const float*)d_in[0];
    const int*   ei = (const int*)d_in[1];
    const float* W1 = (const float*)d_in[2];
    const float* b1 = (const float*)d_in[3];
    const float* W2 = (const float*)d_in[4];
    const float* b2 = (const float*)d_in[5];
    float* out = (float*)d_out;

    const int  N = in_sizes[0] / NF;
    const long E = in_sizes[1] / 2;
    const int* src = ei;
    const int* dst = ei + E;

    float* p    = (float*)d_ws;
    float* deg  = p;                       // N
    float* dinv = p + N;                   // N
    float* h1   = p + 2L * N;              // N*64
    float* out1 = h1 + (long)N * HID;      // N*64
    float* h2   = out1 + (long)N * HID;    // N*16

    const int TB = 256;
    auto blk = [](long t, int b) { return (unsigned)((t + b - 1) / b); };

    // degree + normalization
    deg_init_kernel<<<blk(N, TB), TB, 0, stream>>>(deg, N);
    deg_edge_kernel<<<blk(E, TB), TB, 0, stream>>>(deg, dst, E);
    dinv_kernel<<<blk(N, TB), TB, 0, stream>>>(dinv, deg, N);

    // layer 1
    gemm1_wmma<<<blk(N, 128), 256, 0, stream>>>(x, W1, h1, N);
    agg_init_kernel<HID><<<blk((long)N * HID, TB), TB, 0, stream>>>(out1, h1, dinv, b1, (long)N * HID);
    agg_edge_kernel<HID><<<blk(E * (HID / 4), TB), TB, 0, stream>>>(out1, h1, dinv, src, dst, E);

    // layer 2 (ReLU fused into GEMM2 A-load)
    gemm2_wmma<<<blk(N, 128), 256, 0, stream>>>(out1, W2, h2, N);
    agg_init_kernel<NC><<<blk((long)N * NC, TB), TB, 0, stream>>>(out, h2, dinv, b2, (long)N * NC);
    agg_edge_kernel<NC><<<blk(E * (NC / 4), TB), TB, 0, stream>>>(out, h2, dinv, src, dst, E);
}
